// MatryoshkaSAE_3650722202385
// MI455X (gfx1250) — matrix operand order repaired and checked by reading, not compiled, not run on hardware
//
#include <hip/hip_runtime.h>
#include <hip/hip_bf16.h>

#define N_ROWS 65536
#define D_DIM  128
#define M_DIM  4096
#define K_TOP  64
#define A_PAD  136   // padded LDS row stride (bf16 elems); 272B rows keep 16B align

typedef __attribute__((ext_vector_type(16))) __bf16 v16bf;
typedef __attribute__((ext_vector_type(8)))  float  v8f;

union BV16 {
    v16bf v;
    unsigned short s[16];
    uint4 q[2];
};

__device__ __forceinline__ unsigned short f32_to_bf16_rne(float f) {
    unsigned u = __float_as_uint(f);
    unsigned r = u + 0x7FFFu + ((u >> 16) & 1u);
    return (unsigned short)(r >> 16);
}

// ---------------------------------------------------------------------------
// Pass 0: one-time W_enc f32 -> bf16 into workspace (1 MB).
// ---------------------------------------------------------------------------
__global__ __launch_bounds__(256)
void convert_wenc_bf16(const float* __restrict__ W_enc,
                       unsigned short* __restrict__ Wb) {
    const size_t base = ((size_t)blockIdx.x * 256 + threadIdx.x) * 8;
    union { unsigned short s[8]; uint4 q; } tmp;
#pragma unroll
    for (int i = 0; i < 8; ++i)
        tmp.s[i] = f32_to_bf16_rne(W_enc[base + i]);
    *(uint4*)&Wb[base] = tmp.q;
}

// ---------------------------------------------------------------------------
// Pass 1: z_pre = relu((x - b_dec) @ W_enc^T + b_enc), bf16 WMMA, f32 accum.
// Block: 256 threads = 8 waves; 16 rows of x; each wave does 32 16x16 tiles.
// B panels are double-buffered across tiles so WMMAs overlap the next loads.
// ---------------------------------------------------------------------------
__global__ __launch_bounds__(256)
void sae_encode_wmma(const float* __restrict__ x,
                     const unsigned short* __restrict__ Wb,   // bf16 W_enc [M][D]
                     const float* __restrict__ b_enc,
                     const float* __restrict__ b_dec,
                     float* __restrict__ z) {
    __shared__ unsigned short sA[16 * A_PAD];   // bf16(x - b_dec), 16 rows x 128 K

    const int tid      = threadIdx.x;
    const int row_base = blockIdx.x * 16;

    // Stage A tile: each thread converts 8 consecutive K of one row.
    {
        const int mm = tid >> 4;          // 0..15 row
        const int kb = (tid & 15) << 3;   // 0..120 step 8
        const float* xr = x + (size_t)(row_base + mm) * D_DIM + kb;
        const float* bd = b_dec + kb;
        union { unsigned short s[8]; uint4 q; } tmp;
#pragma unroll
        for (int i = 0; i < 8; ++i)
            tmp.s[i] = f32_to_bf16_rne(xr[i] - bd[i]);
        *(uint4*)&sA[mm * A_PAD + kb] = tmp.q;  // 16B aligned (A_PAD*2 = 272 = 17*16)
    }
    __syncthreads();

    const int lane = tid & 31;
    const int wave = tid >> 5;
    const int n    = lane & 15;   // output column within tile / latent offset
    const int half = lane >> 4;   // lane half per WMMA 16-bit layouts
    const int m    = n;           // A-matrix row owned by this lane

    // Hold all four A fragments (K = 0..127) in registers for the whole loop.
    BV16 a[4];
#pragma unroll
    for (int s = 0; s < 4; ++s) {
        a[s].q[0] = *(const uint4*)&sA[m * A_PAD + s * 32 + half * 16];
        a[s].q[1] = *(const uint4*)&sA[m * A_PAD + s * 32 + half * 16 + 8];
    }

    // This lane's W_enc base row for tile 0; consecutive tiles step 16 rows.
    const unsigned short* wb0 = Wb + (size_t)(wave * 32 * 16 + n) * D_DIM;

    // Preload B panel for tile 0 (8 x b128).
    BV16 bc[4];
#pragma unroll
    for (int s = 0; s < 4; ++s) {
        bc[s].q[0] = *(const uint4*)(wb0 + s * 32 + half * 16);
        bc[s].q[1] = *(const uint4*)(wb0 + s * 32 + half * 16 + 8);
    }

    for (int c = 0; c < 32; ++c) {
        const int j0 = (wave * 32 + c) * 16;

        // Prefetch two tiles ahead into L2 (global_prefetch_b8).
        __builtin_prefetch(Wb + (size_t)(((j0 + 32) & (M_DIM - 1)) + n) * D_DIM, 0, 3);

        // Kick off next tile's B panel while this tile's WMMAs run.
        BV16 bn[4];
        if (c < 31) {
            const unsigned short* wbn = wb0 + (size_t)(c + 1) * 16 * D_DIM;
#pragma unroll
            for (int s = 0; s < 4; ++s) {
                bn[s].q[0] = *(const uint4*)(wbn + s * 32 + half * 16);
                bn[s].q[1] = *(const uint4*)(wbn + s * 32 + half * 16 + 8);
            }
        }

        v8f acc = {0.f, 0.f, 0.f, 0.f, 0.f, 0.f, 0.f, 0.f};
#pragma unroll
        for (int s = 0; s < 4; ++s)
            acc = __builtin_amdgcn_wmma_f32_16x16x32_bf16(
                false, a[s].v, false, bc[s].v, (short)0, acc, false, false);

        // Epilogue: + b_enc, relu, store.  C/D layout: VGPR r -> row (r + 8*half), col n.
        const float be = b_enc[j0 + n];
        float* zp = z + (size_t)row_base * M_DIM + j0 + n;
#pragma unroll
        for (int r = 0; r < 8; ++r) {
            const int mo = r + 8 * half;
            float v = acc[r] + be;
            zp[(size_t)mo * M_DIM] = v > 0.f ? v : 0.f;
        }

        if (c < 31) {
#pragma unroll
            for (int s = 0; s < 4; ++s)
                bc[s] = bn[s];
        }
    }
}

// ---------------------------------------------------------------------------
// Pass 2: per-row top-K (radix select on f32 bits, exact index-order ties),
// sparse write-back of z, and sparse decode x_hat = z @ W_dec^T + b_dec.
// One wave per row; 2 rows per 64-thread block. Row staged into LDS with
// gfx1250 async-to-LDS copies (ASYNCcnt-tracked, no VGPR round trip).
// ---------------------------------------------------------------------------
__global__ __launch_bounds__(64)
void sae_topk_decode(const float* __restrict__ W_dec,
                     const float* __restrict__ b_dec,
                     float* __restrict__ z,        // in: z_pre, out: sparse z
                     float* __restrict__ x_hat) {
    __shared__ float srow[2][M_DIM];   // 32 KB
    __shared__ int   s_j[2][K_TOP];
    __shared__ float s_v[2][K_TOP];

    const int lane = threadIdx.x & 31;
    const int w    = threadIdx.x >> 5;
    const int row  = blockIdx.x * 2 + w;

    float* zrow = z + (size_t)row * M_DIM;

    // Async copy row -> LDS: 8 x GLOBAL_LOAD_ASYNC_TO_LDS_B128 per lane.
    // Generic address low 32 bits == LDS byte offset (aperture mapping).
    {
        const unsigned lds_base = (unsigned)(size_t)(&srow[w][0]);
#pragma unroll
        for (int t = 0; t < 8; ++t) {
            const unsigned dst = lds_base + (unsigned)((t * 128 + lane * 4) * 4);
            const float*   src = zrow + t * 128 + lane * 4;
            asm volatile("global_load_async_to_lds_b128 %0, %1, off"
                         :: "v"(dst), "v"(src) : "memory");
        }
        asm volatile("s_wait_asynccnt 0" ::: "memory");
    }
    __syncthreads();

    // Radix select the K-th largest (values >= 0 so uint order == float order).
    unsigned prefix = 0u;
    for (int bit = 30; bit >= 0; --bit) {
        const unsigned cand = prefix | (1u << bit);
        int cnt = 0;
        for (int t = 0; t < 128; ++t)
            cnt += (__float_as_uint(srow[w][t * 32 + lane]) >= cand) ? 1 : 0;
        for (int off = 16; off > 0; off >>= 1)
            cnt += __shfl_xor(cnt, off, 32);
        if (cnt >= K_TOP) prefix = cand;
    }

    // Count strictly-greater to size the tie budget.
    int cg = 0;
    for (int t = 0; t < 128; ++t)
        cg += (__float_as_uint(srow[w][t * 32 + lane]) > prefix) ? 1 : 0;
    for (int off = 16; off > 0; off >>= 1)
        cg += __shfl_xor(cg, off, 32);
    const int need_eq = K_TOP - cg;

    // Keep pass: exact K entries, ties broken by lowest index (matches top_k).
    const unsigned lt_mask = (1u << lane) - 1u;
    int base_eq = 0, base_keep = 0;
    for (int t = 0; t < 128; ++t) {
        const int j = t * 32 + lane;
        const float v = srow[w][j];
        const unsigned uv = __float_as_uint(v);
        const bool gt = uv > prefix;
        const bool eq = uv == prefix;
        const unsigned eqm = (unsigned)__ballot(eq);
        const int eqrank = base_eq + __popc(eqm & lt_mask);
        base_eq += __popc(eqm);
        const bool keep = gt || (eq && eqrank < need_eq);
        const unsigned km = (unsigned)__ballot(keep);
        const int pos = base_keep + __popc(km & lt_mask);
        base_keep += __popc(km);
        if (keep && pos < K_TOP) { s_j[w][pos] = j; s_v[w][pos] = v; }
        zrow[j] = keep ? v : 0.0f;                 // sparse z write-back
    }
    __syncthreads();

    // Sparse decode: x_hat[row, d] = sum_p v_p * W_dec[d, j_p] + b_dec[d].
    const int d0 = lane * 4;
    float a0 = 0.f, a1 = 0.f, a2 = 0.f, a3 = 0.f;
    for (int p = 0; p < K_TOP; ++p) {
        const int   j = s_j[w][p];
        const float v = s_v[w][p];
        const float* wc = W_dec + j;
        a0 += v * wc[(size_t)(d0 + 0) * M_DIM];
        a1 += v * wc[(size_t)(d0 + 1) * M_DIM];
        a2 += v * wc[(size_t)(d0 + 2) * M_DIM];
        a3 += v * wc[(size_t)(d0 + 3) * M_DIM];
    }
    float4 o;
    o.x = a0 + b_dec[d0 + 0];
    o.y = a1 + b_dec[d0 + 1];
    o.z = a2 + b_dec[d0 + 2];
    o.w = a3 + b_dec[d0 + 3];
    *(float4*)&x_hat[(size_t)row * D_DIM + d0] = o;
}

// ---------------------------------------------------------------------------
extern "C" void kernel_launch(void* const* d_in, const int* in_sizes, int n_in,
                              void* d_out, int out_size, void* d_ws, size_t ws_size,
                              hipStream_t stream) {
    const float* x     = (const float*)d_in[0];
    const float* W_enc = (const float*)d_in[1];
    const float* b_enc = (const float*)d_in[2];
    const float* W_dec = (const float*)d_in[3];
    const float* b_dec = (const float*)d_in[4];

    float* x_hat = (float*)d_out;                               // [N, D]
    float* z     = (float*)d_out + (size_t)N_ROWS * D_DIM;      // [N, M]
    unsigned short* Wb = (unsigned short*)d_ws;                 // bf16 W_enc, 1 MB

    // Pass 0: one-time weight conversion to bf16.
    convert_wenc_bf16<<<(M_DIM * D_DIM) / (256 * 8), 256, 0, stream>>>(W_enc, Wb);
    // Pass 1: dense encode GEMM (bf16 WMMA), z_pre -> z region.
    sae_encode_wmma<<<N_ROWS / 16, 256, 0, stream>>>(x, Wb, b_enc, b_dec, z);
    // Pass 2: top-K sparsify in place + sparse decode.
    sae_topk_decode<<<N_ROWS / 2, 64, 0, stream>>>(W_dec, b_dec, z, x_hat);
}
